// PixelPro_20418274525351
// MI455X (gfx1250) — matrix operand order repaired and checked by reading, not compiled, not run on hardware
//
#include <hip/hip_runtime.h>
#include <hip/hip_bf16.h>

typedef __bf16 bf16_t;
typedef __attribute__((ext_vector_type(16))) __bf16 v16bf;
typedef __attribute__((ext_vector_type(8)))  __bf16 v8bf;
typedef __attribute__((ext_vector_type(8)))  float   v8f;
typedef __attribute__((ext_vector_type(4)))  unsigned int v4u;
typedef __attribute__((ext_vector_type(8)))  int v8i;
typedef __attribute__((ext_vector_type(4)))  int v4i;

#define LDS_AS __attribute__((address_space(3)))
typedef LDS_AS bf16_t       lds_bf16;
typedef LDS_AS const bf16_t lds_cbf16;

#define B_DIM 4
#define C_DIM 256
#define N_DIM 3136   // 56*56 = 49 * 64
#define JT 64        // output-column tile per block
#define IT 64        // affinity row tile per iteration
#define LDP 88       // LDS row stride (elems) for P: 176B, 16B-aligned, conflict-free
#define XI_LD 264    // LDS row stride for XnI tile: 528B (512B row + 16B TDM pad)
#define GT_LD 72     // LDS row stride for G tile: 144B (128B row + 16B TDM pad)
#define XI_SZ (64 * XI_LD)    // elements per XnI buffer
#define GT_SZ (256 * GT_LD)   // elements per G buffer

#define SMEM_BYTES ((2 * XI_SZ + 2 * GT_SZ + 64 * LDP) * 2)

#if defined(__has_builtin)
#if __has_builtin(__builtin_amdgcn_tensor_load_to_lds) && \
    __has_builtin(__builtin_amdgcn_s_wait_tensorcnt)
#define HAVE_TDM 1
#endif
#if __has_builtin(__builtin_amdgcn_sched_barrier)
#define SCHED_FENCE() __builtin_amdgcn_sched_barrier(0)
#endif
#endif
#ifndef HAVE_TDM
#define HAVE_TDM 0
#endif
#ifndef SCHED_FENCE
#define SCHED_FENCE()
#endif

// ---- fragment helpers (wave32, v_wmma_f32_16x16x32_bf16 layouts) ----------
// A (16x32, MxK): lane<16 holds row M=lane, K = k..k+15;
//                 lane>=16 holds row M=lane-16, K = k+16..k+31.
// B (32x16, KxN) read from [n][k] storage:
// lane<16 holds col N=lane with K = k..k+15; lane>=16 holds K = k+16..k+31.
__device__ __forceinline__ v16bf load_b_frag_g(const bf16_t* row, int k, int hi) {
    const bf16_t* p = row + k + hi * 16;
    v8bf lo = *(const v8bf*)(p);
    v8bf hh = *(const v8bf*)(p + 8);
    return __builtin_shufflevector(lo, hh, 0, 1, 2, 3, 4, 5, 6, 7,
                                           8, 9, 10, 11, 12, 13, 14, 15);
}
__device__ __forceinline__ v16bf lds_load_a_frag(lds_cbf16* row, int k, int hi) {
    v8bf lo = *(LDS_AS const v8bf*)(row + k + hi * 16);
    v8bf hh = *(LDS_AS const v8bf*)(row + k + hi * 16 + 8);
    return __builtin_shufflevector(lo, hh, 0, 1, 2, 3, 4, 5, 6, 7,
                                           8, 9, 10, 11, 12, 13, 14, 15);
}
__device__ __forceinline__ v16bf lds_load_b_frag(lds_cbf16* row, int k, int hi) {
    lds_cbf16* p = row + k + hi * 16;
    v8bf lo = *(LDS_AS const v8bf*)(p);
    v8bf hh = *(LDS_AS const v8bf*)(p + 8);
    return __builtin_shufflevector(lo, hh, 0, 1, 2, 3, 4, 5, 6, 7,
                                           8, 9, 10, 11, 12, 13, 14, 15);
}
__device__ __forceinline__ v8f wmma_bf16(v16bf a, v16bf b, v8f c) {
    return __builtin_amdgcn_wmma_f32_16x16x32_bf16(
        false, a, false, b, (short)0, c, false, false);
}

#if HAVE_TDM
__device__ __forceinline__ unsigned lds_addr_of(lds_cbf16* p) {
    return (unsigned)(unsigned long long)p;   // ptrtoint of AS(3) = LDS offset
}
// Issue one 2-D TDM tile load (bf16 elements, data_size code = 1).
// pad_int encodes 2^(pad_int+1) dwords between pads; pad_amt = (dwords-1).
__device__ __forceinline__ void tdm_load_tile_2d(
    const bf16_t* gsrc, unsigned lds_byte,
    unsigned tensor_d0, unsigned tensor_d1, unsigned stride_d0,
    unsigned tile_d0, unsigned tile_d1,
    unsigned pad_int, unsigned pad_amt) {
    unsigned long long ga = (unsigned long long)gsrc;
    v4u g0;
    g0[0] = 1u;                                    // count=1, user descriptor
    g0[1] = lds_byte;                              // lds_addr
    g0[2] = (unsigned)(ga & 0xffffffffu);          // global_addr lo
    g0[3] = (unsigned)((ga >> 32) & 0x01ffffffu)   // global_addr hi [56:32]
            | 0x80000000u;                         // type = 2 ("image")
    v8i g1;
    g1[0] = (int)((1u << 16)                       // data_size = 1 (2 bytes)
                  | (1u << 20)                     // pad_enable
                  | (pad_int << 22) | (pad_amt << 25));
    g1[1] = (int)((tensor_d0 & 0xffffu) << 16);    // tensor_dim0 lo16
    g1[2] = (int)(((tensor_d0 >> 16) & 0xffffu)    // tensor_dim0 hi16
                  | ((tensor_d1 & 0xffffu) << 16));// tensor_dim1 lo16
    g1[3] = (int)(((tensor_d1 >> 16) & 0xffffu)    // tensor_dim1 hi16
                  | (tile_d0 << 16));              // tile_dim0
    g1[4] = (int)(tile_d1 & 0xffffu);              // tile_dim1 (tile_dim2=0)
    g1[5] = (int)stride_d0;                        // tensor_dim0_stride lo32
    g1[6] = 0;
    g1[7] = 0;
    v4i z4 = {0, 0, 0, 0};
#if __clang_major__ >= 23
    v8i z8 = {0, 0, 0, 0, 0, 0, 0, 0};
    __builtin_amdgcn_tensor_load_to_lds(g0, g1, z4, z4, z8, 0);
#else
    __builtin_amdgcn_tensor_load_to_lds(g0, g1, z4, z4, 0);
#endif
}
#endif  // HAVE_TDM

// ---- kernel 1: per-pixel L2 normalize, write xn^T as bf16 [B][N][C] -------
__global__ void prep_norm_t(const float* __restrict__ x,
                            bf16_t* __restrict__ xnT) {
    int t = blockIdx.x * blockDim.x + threadIdx.x;   // exact grid: B*N threads
    int b = t / N_DIM;
    int n = t - b * N_DIM;
    const float* xp = x + (size_t)b * C_DIM * N_DIM + n;
    float ss = 0.f;
#pragma unroll 8
    for (int c = 0; c < C_DIM; ++c) {
        float v = xp[(size_t)c * N_DIM];
        ss = fmaf(v, v, ss);
    }
    float inv = 1.0f / fmaxf(sqrtf(ss), 1e-12f);
    bf16_t* dst = xnT + ((size_t)b * N_DIM + n) * C_DIM;
    for (int c = 0; c < C_DIM; c += 8) {
        v8bf pk;
#pragma unroll
        for (int u = 0; u < 8; ++u)
            pk[u] = (bf16_t)(xp[(size_t)(c + u) * N_DIM] * inv);
        *(v8bf*)(dst + c) = pk;
    }
}

// ---- kernel 2: 1x1-conv GEMM g = Wx + b, write bf16 [B][C][N] -------------
__global__ void prep_conv(const float* __restrict__ x,
                          const float* __restrict__ w,
                          const float* __restrict__ bias,
                          bf16_t* __restrict__ g) {
    int n  = blockIdx.x * 256 + threadIdx.x;
    int o0 = blockIdx.y * 8;
    int b  = blockIdx.z;
    if (n >= N_DIM) return;
    const float* xp = x + (size_t)b * C_DIM * N_DIM + n;
    float acc[8];
#pragma unroll
    for (int u = 0; u < 8; ++u) acc[u] = bias[o0 + u];
#pragma unroll 4
    for (int c = 0; c < C_DIM; ++c) {
        float xv = xp[(size_t)c * N_DIM];
#pragma unroll
        for (int u = 0; u < 8; ++u)
            acc[u] = fmaf(w[(size_t)(o0 + u) * C_DIM + c], xv, acc[u]);
    }
#pragma unroll
    for (int u = 0; u < 8; ++u)
        g[((size_t)b * C_DIM + (o0 + u)) * N_DIM + n] = (bf16_t)acc[u];
}

// ---- kernel 3: fused  out[:,J] = sum_I  G[:,I] * relu(XnI^T XnJ)^2 --------
__global__ __launch_bounds__(256)
void fused_affinity(const bf16_t* __restrict__ xnT,
                    const bf16_t* __restrict__ g,
                    float* __restrict__ out) {
    extern __shared__ __align__(16) char smem_raw[];
    lds_bf16* smem = (lds_bf16*)smem_raw;        // single addrspacecast to AS(3)
    lds_bf16* Xi0 = smem;                        // [2][64 x XI_LD]
    lds_bf16* Gt0 = smem + 2 * XI_SZ;            // [2][256 x GT_LD]
    lds_bf16* Pl  = smem + 2 * XI_SZ + 2 * GT_SZ;// [64 x LDP]

    const int j0   = blockIdx.x * JT;
    const int b    = blockIdx.y;
    const int tid  = threadIdx.x;
    const int wave = tid >> 5;
    const int lane = tid & 31;
    const int l15  = lane & 15;
    const int hi   = lane >> 4;

    const bf16_t* Xb = xnT + (size_t)b * N_DIM * C_DIM;
    const bf16_t* Gb = g   + (size_t)b * C_DIM * N_DIM;

    const int ti = wave >> 1;          // stage-1 i tile (0..3)
    const int tj = (wave & 1) * 2;     // stage-1 j tiles {tj, tj+1}

    // Loop-invariant XnJ B-fragments: 8 k-steps x 2 tiles resident in VGPRs.
    v16bf bj0[8], bj1[8];
    {
        const bf16_t* Brow0 = Xb + (size_t)(j0 + tj * 16 + l15) * C_DIM;
        const bf16_t* Brow1 = Xb + (size_t)(j0 + (tj + 1) * 16 + l15) * C_DIM;
#pragma unroll
        for (int k8 = 0; k8 < 8; ++k8) {
            bj0[k8] = load_b_frag_g(Brow0, k8 * 32, hi);
            bj1[k8] = load_b_frag_g(Brow1, k8 * 32, hi);
        }
    }

    v8f acc[2][4];
#pragma unroll
    for (int rt = 0; rt < 2; ++rt)
#pragma unroll
        for (int ct = 0; ct < 4; ++ct)
#pragma unroll
            for (int r = 0; r < 8; ++r) acc[rt][ct][r] = 0.0f;

    int buf = 0;
#if HAVE_TDM
    if (wave == 0) {
        // XnI tile: 64 rows x 256 el, row = 512B -> pad 16B every 128 dwords
        tdm_load_tile_2d(Xb, lds_addr_of(Xi0),
                         C_DIM, N_DIM, C_DIM, C_DIM, IT, 6, 3);
        // G tile: 256 rows x 64 el, row = 128B -> pad 16B every 32 dwords
        tdm_load_tile_2d(Gb, lds_addr_of(Gt0),
                         N_DIM, C_DIM, N_DIM, IT, C_DIM, 4, 3);
        __builtin_amdgcn_s_wait_tensorcnt((short)0);
    }
    __syncthreads();
#endif

    for (int i0 = 0; i0 < N_DIM; i0 += IT) {
        lds_cbf16* XiBuf = Xi0 + buf * XI_SZ;
        lds_cbf16* GtBuf = Gt0 + buf * GT_SZ;
#if HAVE_TDM
        if (wave == 0 && (i0 + IT) < N_DIM) {   // async-prefetch next tiles
            tdm_load_tile_2d(Xb + (size_t)(i0 + IT) * C_DIM,
                             lds_addr_of(Xi0 + (buf ^ 1) * XI_SZ),
                             C_DIM, N_DIM - (i0 + IT), C_DIM, C_DIM, IT, 6, 3);
            tdm_load_tile_2d(Gb + (i0 + IT),
                             lds_addr_of(Gt0 + (buf ^ 1) * GT_SZ),
                             N_DIM - (i0 + IT), C_DIM, N_DIM, IT, C_DIM, 4, 3);
        }
#else
        // cooperative global->LDS staging (same padded layout), ds stores
        for (int ch = tid; ch < 64 * 32; ch += 256) {
            int r = ch >> 5, c = (ch & 31) << 3;
            v8bf v = *(const v8bf*)(Xb + (size_t)(i0 + r) * C_DIM + c);
            *(LDS_AS v8bf*)(Xi0 + buf * XI_SZ + r * XI_LD + c) = v;
        }
        for (int ch = tid; ch < 256 * 8; ch += 256) {
            int r = ch >> 3, c = (ch & 7) << 3;
            v8bf v = *(const v8bf*)(Gb + (size_t)r * N_DIM + i0 + c);
            *(LDS_AS v8bf*)(Gt0 + buf * GT_SZ + r * GT_LD + c) = v;
        }
        __syncthreads();
#endif
        // ---- stage 1: S = XnI^T * XnJ (K=256) -----------------------------
        // All 16 ds_loads issued first; sched_barrier(0) pins them above the
        // WMMA burst so waits become s_wait_dscnt N>0 (LDS returns in order).
        lds_cbf16* ArowL = XiBuf + (ti * 16 + l15) * XI_LD;
        v16bf afr[8];
#pragma unroll
        for (int k8 = 0; k8 < 8; ++k8)
            afr[k8] = lds_load_a_frag(ArowL, k8 * 32, hi);
        SCHED_FENCE();
        v8f s0, s1;
#pragma unroll
        for (int r = 0; r < 8; ++r) { s0[r] = 0.0f; s1[r] = 0.0f; }
#pragma unroll
        for (int k8 = 0; k8 < 8; ++k8) {
            s0 = wmma_bf16(afr[k8], bj0[k8], s0);
            s1 = wmma_bf16(afr[k8], bj1[k8], s1);
        }
        // relu^2 -> bf16; D-frag rows are contiguous: one ds_store_b128/tile
        {
            int ibase = ti * 16 + hi * 8;
            v8bf p0, p1;
#pragma unroll
            for (int r = 0; r < 8; ++r) {
                float t0 = fmaxf(s0[r], 0.0f);
                float t1 = fmaxf(s1[r], 0.0f);
                p0[r] = (bf16_t)(t0 * t0);
                p1[r] = (bf16_t)(t1 * t1);
            }
            *(LDS_AS v8bf*)(Pl + (tj * 16 + l15) * LDP + ibase)       = p0;
            *(LDS_AS v8bf*)(Pl + ((tj + 1) * 16 + l15) * LDP + ibase) = p1;
        }
        __syncthreads();

        // ---- stage 2: acc += G[:,I] * P (K=64): 12 loads, fence, 16 WMMAs -
        {
            v16bf bp[2][4];   // [kx][ct]
            v16bf ag[2][2];   // [kx][rt]
#pragma unroll
            for (int kx = 0; kx < 2; ++kx) {
#pragma unroll
                for (int ct = 0; ct < 4; ++ct)
                    bp[kx][ct] = lds_load_b_frag(
                        Pl + (ct * 16 + l15) * LDP, kx * 32, hi);
#pragma unroll
                for (int rt = 0; rt < 2; ++rt)
                    ag[kx][rt] = lds_load_a_frag(
                        GtBuf + (wave * 32 + rt * 16 + l15) * GT_LD,
                        kx * 32, hi);
            }
            SCHED_FENCE();
#pragma unroll
            for (int kx = 0; kx < 2; ++kx)
#pragma unroll
                for (int rt = 0; rt < 2; ++rt)
#pragma unroll
                    for (int ct = 0; ct < 4; ++ct)
                        acc[rt][ct] =
                            wmma_bf16(ag[kx][rt], bp[kx][ct], acc[rt][ct]);
        }
#if HAVE_TDM
        if (wave == 0) __builtin_amdgcn_s_wait_tensorcnt((short)0);
        __syncthreads();   // P consumed + next buffers resident
        buf ^= 1;
#else
        __syncthreads();   // P and tiles consumed before next fill
#endif
    }

    // ---- epilogue: write f32 output tile ----------------------------------
    float* Ob = out + (size_t)b * C_DIM * N_DIM;
#pragma unroll
    for (int rt = 0; rt < 2; ++rt) {
        int cbase = wave * 32 + rt * 16 + hi * 8;
#pragma unroll
        for (int ct = 0; ct < 4; ++ct) {
            int n = j0 + ct * 16 + l15;
#pragma unroll
            for (int r = 0; r < 8; ++r)
                Ob[(size_t)(cbase + r) * N_DIM + n] = acc[rt][ct][r];
        }
    }
}

// ---------------------------------------------------------------------------
extern "C" void kernel_launch(void* const* d_in, const int* in_sizes, int n_in,
                              void* d_out, int out_size, void* d_ws, size_t ws_size,
                              hipStream_t stream) {
    const float* x    = (const float*)d_in[0];   // [4,256,56,56]
    const float* w    = (const float*)d_in[1];   // [256,256,1,1]
    const float* bias = (const float*)d_in[2];   // [256]
    float* out = (float*)d_out;                  // [4,256,56,56] f32

    bf16_t* xnT = (bf16_t*)d_ws;                           // [B][N][C] bf16
    bf16_t* g   = xnT + (size_t)B_DIM * N_DIM * C_DIM;     // [B][C][N] bf16

    prep_norm_t<<<dim3((B_DIM * N_DIM) / 256), 256, 0, stream>>>(x, xnT);
    prep_conv<<<dim3((N_DIM + 255) / 256, C_DIM / 8, B_DIM), 256, 0, stream>>>(
        x, w, bias, g);
    fused_affinity<<<dim3(N_DIM / JT, B_DIM), 256, SMEM_BYTES, stream>>>(
        xnT, g, out);
}
